// MambaSSM_10831907520986
// MI455X (gfx1250) — matrix operand order repaired and checked by reading, not compiled, not run on hardware
//
#include <hip/hip_runtime.h>
#include <hip/hip_bf16.h>

typedef unsigned short u16;
typedef __bf16 bf16_t;
typedef __attribute__((ext_vector_type(8)))  bf16_t v8bf;
typedef __attribute__((ext_vector_type(16))) bf16_t v16bf;
typedef __attribute__((ext_vector_type(8)))  float  v8f;

#define D_MODEL 1024
#define D_STATE 16
#define D_CONV  4
#define D_INNER 2048
#define B_SZ    4
#define SEQ_L   2048
#define ROWS    (B_SZ * SEQ_L)          /* 8192 */

__device__ __forceinline__ u16 f2bf(float f) {
    unsigned int u = __float_as_uint(f);
    u += 0x7FFFu + ((u >> 16) & 1u);    /* round to nearest even */
    return (u16)(u >> 16);
}

/* CDNA5 async copy: global -> LDS, 16B per lane, tracked by ASYNCcnt.
 * Low 32 bits of a flat __shared__ pointer are the LDS byte offset
 * (flat LDS aperture). */
__device__ __forceinline__ void async_copy_b128(unsigned lds_addr, const void* gptr) {
    asm volatile("global_load_async_to_lds_b128 %0, %1, off"
                 :: "v"(lds_addr), "v"(gptr) : "memory");
}
__device__ __forceinline__ void wait_async0() {
    asm volatile("s_wait_asynccnt 0" ::: "memory");
}
/* allow 2 outstanding async ops (the tile-(it+2) pair) while guaranteeing
 * everything older has landed — ASYNCcnt completes in order per wave */
__device__ __forceinline__ void wait_async2() {
    asm volatile("s_wait_asynccnt 2" ::: "memory");
}
__device__ __forceinline__ unsigned lds_off(const void* p) {
    return (unsigned)(unsigned long long)p;
}

/* ---------------- fp32 -> bf16 conversion ---------------- */
__global__ __launch_bounds__(256)
void k_cvt_bf16(const float* __restrict__ src, u16* __restrict__ dst, int n) {
    int i = blockIdx.x * 256 + threadIdx.x;
    int stride = gridDim.x * 256;
    for (; i < n; i += stride) dst[i] = f2bf(src[i]);
}

/* ---------------- bf16 WMMA GEMM: C(f32) = A(bf16,MxK) @ B(bf16,KxN) ----------------
 * Block tile 128x128, K-step 32, 8 waves (2x4), wave tile 64x32 = 4x2 WMMA tiles.
 * Triple-buffered LDS: A tiles staged via GLOBAL_LOAD_ASYNC_TO_LDS_B128 two K-steps
 * ahead (ASYNCcnt, partial wait s_wait_asynccnt 2); B tile register-staged with a
 * 4x4 micro-tile transpose so LDS writes are ds_store_b64 at full bank width.
 * Fragment reads are 16B-aligned ds_load_b128 (pitch 40 halves = 80B rows).
 */
#define BM 128
#define BN 128
#define BKK 32
#define LP 40

template <int N, int K>
__global__ __launch_bounds__(256)
void k_gemm_bf16(const u16* __restrict__ A, const u16* __restrict__ Bm,
                 float* __restrict__ C) {
    __shared__ u16 As[3][BM * LP];
    __shared__ u16 Bs[3][BN * LP];

    const int tid  = threadIdx.x;
    const int lane = tid & 31;
    const int wave = tid >> 5;
    const int wrow = wave >> 2;          /* 0..1 */
    const int wcol = wave & 3;           /* 0..3 */
    const int bm = blockIdx.y * BM;
    const int bn = blockIdx.x * BN;

    const int mlane = lane & 15;
    const int hi16  = lane >> 4;         /* 0 or 1 */
    const int kbA   = hi16 * 8;          /* A frag K base 0/8 (+16 upper half)  */
    const int kbB   = hi16 * 8;          /* B frag K base 0/8 (+16 upper half)  */

    /* A staging: thread -> (row, 16-half chunk) of the 128x32 tile */
    const int am = tid >> 1;
    const int ak = (tid & 1) * 16;
    const u16* gA = A + (size_t)(bm + am) * K + ak;

    /* B staging: thread -> 4(k) x 4(n) micro-tile of the 32x128 tile */
    const int bkq = (tid >> 5) * 4;      /* k quad base 0,4,..,28 */
    const int bnb = (tid & 31) * 4;      /* n base 0..124         */
    const u16* gB = Bm + (size_t)bkq * N + bn + bnb;

    v8f acc[4][2];
#pragma unroll
    for (int mi = 0; mi < 4; ++mi)
#pragma unroll
        for (int ni = 0; ni < 2; ++ni)
#pragma unroll
            for (int r = 0; r < 8; ++r) acc[mi][ni][r] = 0.f;

    const int nT = K / BKK;

    u16 r0[4], r1[4], r2[4], r3[4];      /* B micro-tile registers */

    auto loadB = [&](int it) {
        const u16* gb = gB + (size_t)it * BKK * N;
        *(uint2*)r0 = *(const uint2*)gb;
        *(uint2*)r1 = *(const uint2*)(gb + N);
        *(uint2*)r2 = *(const uint2*)(gb + 2 * N);
        *(uint2*)r3 = *(const uint2*)(gb + 3 * N);
    };
    auto storeB = [&](int buf) {
#pragma unroll
        for (int i = 0; i < 4; ++i) {
            uint2 v;
            v.x = (unsigned)r0[i] | ((unsigned)r1[i] << 16);   /* k, k+1 */
            v.y = (unsigned)r2[i] | ((unsigned)r3[i] << 16);   /* k+2, k+3 */
            *(uint2*)&Bs[buf][(bnb + i) * LP + bkq] = v;       /* ds_store_b64 */
        }
    };
    auto stageA = [&](int it, int buf) {
        const u16* ga = gA + (size_t)it * BKK;
        async_copy_b128(lds_off(&As[buf][am * LP + ak]), ga);
        async_copy_b128(lds_off(&As[buf][am * LP + ak + 8]), ga + 8);
    };

    /* ---- prologue: A tiles 0,1 in flight; B tile 0 resident ---- */
    stageA(0, 0);
    if (nT > 1) stageA(1, 1);
    loadB(0);
    storeB(0);
    if (nT > 1) wait_async2(); else wait_async0();
    __syncthreads();

    int cur = 0;
    for (int it = 0; it < nT; ++it) {
        int b1 = cur + 1; if (b1 >= 3) b1 -= 3;
        int b2 = cur + 2; if (b2 >= 3) b2 -= 3;
        const bool moreB = (it + 1) < nT;
        const bool moreA = (it + 2) < nT;

        /* issue tile it+2's async A copies; pull tile it+1's B into regs */
        if (moreA) stageA(it + 2, b2);
        if (moreB) loadB(it + 1);

        /* ---- fragments + 8 WMMAs from current buffer ---- */
        v16bf af[4], bfr[2];
#pragma unroll
        for (int mi = 0; mi < 4; ++mi) {
            const u16* p = &As[cur][(wrow * 64 + mi * 16 + mlane) * LP + kbA];
            v8bf lo = *(const v8bf*)p;           /* K kb..kb+7   */
            v8bf hi = *(const v8bf*)(p + 16);    /* K kb+16..+23 */
            af[mi] = __builtin_shufflevector(lo, hi, 0,1,2,3,4,5,6,7,8,9,10,11,12,13,14,15);
        }
#pragma unroll
        for (int ni = 0; ni < 2; ++ni) {
            const u16* p = &Bs[cur][(wcol * 32 + ni * 16 + mlane) * LP + kbB];
            v8bf lo = *(const v8bf*)p;
            v8bf hi = *(const v8bf*)(p + 16);
            bfr[ni] = __builtin_shufflevector(lo, hi, 0,1,2,3,4,5,6,7,8,9,10,11,12,13,14,15);
        }
#pragma unroll
        for (int mi = 0; mi < 4; ++mi)
#pragma unroll
            for (int ni = 0; ni < 2; ++ni)
                acc[mi][ni] = __builtin_amdgcn_wmma_f32_16x16x32_bf16(
                    false, af[mi], false, bfr[ni], (short)0, acc[mi][ni], false, false);

        /* ---- finish staging tile it+1; tolerate tile it+2 in flight ---- */
        if (moreB) storeB(b1);
        if (moreA) wait_async2(); else wait_async0();
        __syncthreads();
        cur = b1;
    }

    /* store: VGPR r -> M = r (lanes 0-15) / r+8 (lanes 16-31), N = lane&15 */
#pragma unroll
    for (int mi = 0; mi < 4; ++mi)
#pragma unroll
        for (int ni = 0; ni < 2; ++ni) {
            int row0 = bm + wrow * 64 + mi * 16 + hi16 * 8;
            int col  = bn + wcol * 32 + ni * 16 + mlane;
            float* cp = C + (size_t)row0 * N + col;
#pragma unroll
            for (int r = 0; r < 8; ++r) cp[(size_t)r * N] = acc[mi][ni][r];
        }
}

/* ---------------- causal depthwise conv (K=4) + SiLU ---------------- */
__global__ __launch_bounds__(256)
void k_conv_silu(const float* __restrict__ xz, const float* __restrict__ conv_w,
                 const float* __restrict__ conv_b, float* __restrict__ xc) {
    int i = blockIdx.x * 256 + threadIdx.x;          /* over ROWS*D_INNER */
    if (i >= ROWS * D_INNER) return;
    int d   = i & (D_INNER - 1);
    int row = i >> 11;
    int l   = row & (SEQ_L - 1);
    int b   = row >> 11;
    float s = conv_b[d];
#pragma unroll
    for (int k = 0; k < D_CONV; ++k) {
        int li = l + k - (D_CONV - 1);
        if (li >= 0)
            s += xz[((size_t)(b * SEQ_L + li)) * (2 * D_INNER) + d] * conv_w[d * D_CONV + k];
    }
    xc[i] = s / (1.f + __expf(-s));
}

/* ---------------- ssm_in = xc @ W_xproj -> Bc, Cc, dt_raw ---------------- */
__global__ __launch_bounds__(256)
void k_xproj(const float* __restrict__ xc, const float* __restrict__ Wx,
             float* __restrict__ Bc, float* __restrict__ Cc, float* __restrict__ dtr) {
    int w    = blockIdx.x * 8 + (threadIdx.x >> 5);
    int lane = threadIdx.x & 31;
    int row  = w / 33;
    int j    = w - row * 33;
    if (row >= ROWS) return;
    const float* xr = xc + (size_t)row * D_INNER;
    float acc = 0.f;
    for (int k = lane; k < D_INNER; k += 32)
        acc += xr[k] * Wx[(size_t)k * 33 + j];
#pragma unroll
    for (int m = 16; m >= 1; m >>= 1) acc += __shfl_xor(acc, m, 32);
    if (lane == 0) {
        if (j < 16)       Bc[row * 16 + j]        = acc;
        else if (j < 32)  Cc[row * 16 + (j - 16)] = acc;
        else              dtr[row]                = acc;
    }
}

/* ---------------- selective scan: one thread per (b, d, n) ---------------- */
__global__ __launch_bounds__(256)
void k_scan(const float* __restrict__ xc, const float* __restrict__ dtr,
            const float* __restrict__ Bc, const float* __restrict__ Cc,
            const float* __restrict__ A_log, const float* __restrict__ W_dt,
            const float* __restrict__ b_dt, float* __restrict__ y) {
    int t = blockIdx.x * 256 + threadIdx.x;          /* B_SZ*D_INNER*16 */
    int n = t & 15;
    int d = (t >> 4) & (D_INNER - 1);
    int b = t >> 15;
    float a   = -__expf(A_log[d * D_STATE + n]);
    float wdt = W_dt[d];
    float bdt = b_dt[d];
    float h = 0.f;
    for (int l = 0; l < SEQ_L; ++l) {
        size_t row = (size_t)b * SEQ_L + l;
        float xv = xc[row * D_INNER + d];
        float dr = dtr[row];
        float Bn = Bc[row * 16 + n];
        float Cn = Cc[row * 16 + n];
        float v  = dr * wdt + bdt;
        float dt = (v > 20.f) ? v : log1pf(__expf(v));
        h = h * __expf(dt * a) + (dt * xv) * Bn;
        float c = h * Cn;
        c += __shfl_xor(c, 1, 32);
        c += __shfl_xor(c, 2, 32);
        c += __shfl_xor(c, 4, 32);
        c += __shfl_xor(c, 8, 32);
        if (n == 0) y[row * D_INNER + d] = c;
    }
}

/* ---------------- (y + xc*D) * silu(z) -> bf16 for output GEMM ---------------- */
__global__ __launch_bounds__(256)
void k_combine(const float* __restrict__ y, const float* __restrict__ xc,
               const float* __restrict__ xz, const float* __restrict__ Dp,
               u16* __restrict__ ybf) {
    int i = blockIdx.x * 256 + threadIdx.x;
    if (i >= ROWS * D_INNER) return;
    int d   = i & (D_INNER - 1);
    int row = i >> 11;
    float z  = xz[(size_t)row * (2 * D_INNER) + D_INNER + d];
    float sz = z / (1.f + __expf(-z));
    float v  = (y[i] + xc[i] * Dp[d]) * sz;
    ybf[i] = f2bf(v);
}

/* ---------------- host-side orchestration ---------------- */
extern "C" void kernel_launch(void* const* d_in, const int* in_sizes, int n_in,
                              void* d_out, int out_size, void* d_ws, size_t ws_size,
                              hipStream_t stream) {
    const float* x      = (const float*)d_in[0];
    const float* W_in   = (const float*)d_in[1];
    const float* conv_w = (const float*)d_in[2];
    const float* conv_b = (const float*)d_in[3];
    const float* W_xp   = (const float*)d_in[4];
    const float* W_dt   = (const float*)d_in[5];
    const float* b_dt   = (const float*)d_in[6];
    const float* A_log  = (const float*)d_in[7];
    const float* Dp     = (const float*)d_in[8];
    const float* W_out  = (const float*)d_in[9];
    float* out = (float*)d_out;

    char* p = (char*)d_ws;
    auto take = [&p](size_t bytes) -> char* {
        char* r = p;
        p += (bytes + 255) & ~(size_t)255;
        return r;
    };
    const size_t nX    = (size_t)ROWS * D_MODEL;          /* 8.4M  */
    const size_t nWin  = (size_t)D_MODEL * 2 * D_INNER;   /* 4.2M  */
    const size_t nWout = (size_t)D_INNER * D_MODEL;       /* 2.1M  */
    const size_t nXZ   = (size_t)ROWS * 2 * D_INNER;      /* 33.5M */
    const size_t nXC   = (size_t)ROWS * D_INNER;          /* 16.8M */

    u16*   xbf    = (u16*)  take(nX * 2);
    u16*   winbf  = (u16*)  take(nWin * 2);
    u16*   woutbf = (u16*)  take(nWout * 2);
    float* xz     = (float*)take(nXZ * 4);
    float* xc     = (float*)take(nXC * 4);
    float* yb     = (float*)take(nXC * 4);
    u16*   ybf    = (u16*)  take(nXC * 2);
    float* Bcb    = (float*)take((size_t)ROWS * 16 * 4);
    float* Ccb    = (float*)take((size_t)ROWS * 16 * 4);
    float* dtr    = (float*)take((size_t)ROWS * 4);

    /* 1. bf16 casts */
    k_cvt_bf16<<<4096, 256, 0, stream>>>(x, xbf, (int)nX);
    k_cvt_bf16<<<4096, 256, 0, stream>>>(W_in, winbf, (int)nWin);
    k_cvt_bf16<<<4096, 256, 0, stream>>>(W_out, woutbf, (int)nWout);

    /* 2. xz = x @ W_in   (8192 x 1024 x 4096) */
    k_gemm_bf16<2 * D_INNER, D_MODEL>
        <<<dim3((2 * D_INNER) / BN, ROWS / BM), 256, 0, stream>>>(xbf, winbf, xz);

    /* 3. causal conv + SiLU */
    k_conv_silu<<<(ROWS * D_INNER) / 256, 256, 0, stream>>>(xz, conv_w, conv_b, xc);

    /* 4. xproj -> Bc, Cc, dt_raw */
    {
        int waves = ROWS * 33;
        k_xproj<<<(waves + 7) / 8, 256, 0, stream>>>(xc, W_xp, Bcb, Ccb, dtr);
    }

    /* 5. selective scan */
    k_scan<<<(B_SZ * D_INNER * D_STATE) / 256, 256, 0, stream>>>(
        xc, dtr, Bcb, Ccb, A_log, W_dt, b_dt, yb);

    /* 6. combine + cast to bf16 */
    k_combine<<<(ROWS * D_INNER) / 256, 256, 0, stream>>>(yb, xc, xz, Dp, ybf);

    /* 7. out = y @ W_out  (8192 x 2048 x 1024) */
    k_gemm_bf16<D_MODEL, D_INNER>
        <<<dim3(D_MODEL / BN, ROWS / BM), 256, 0, stream>>>(ybf, woutbf, out);
}
